// CNF_3307124818030
// MI455X (gfx1250) — matrix operand order repaired and checked by reading, not compiled, not run on hardware
//
#include <hip/hip_runtime.h>
#include <math.h>

// ---------------------------------------------------------------------------
// Problem constants: B=2048, D=32, H=512. Output is (B, 33): 32 cols dx + div.
// Divergence via exact identity: tr(J) = g1^T (W2 ⊙ S^T) g2, S = W3 @ W1[:32,:]
// ---------------------------------------------------------------------------
#define BATCH 2048
#define DIN   32
#define HID   512
#define MT    16          // batch rows per workgroup
#define PH    520         // LDS pitch (halfwords) for 512-wide buffers (bank-conflict pad, mult of 8)
#define PI    72          // LDS pitch (halfwords) for 64-wide input buffer (mult of 8)

typedef __attribute__((ext_vector_type(16))) __bf16       v16bf;
typedef __attribute__((ext_vector_type(8)))  float        v8f;
typedef __attribute__((ext_vector_type(4)))  unsigned int v4u;

union V16BF { v4u u[2]; v16bf v; };

// gfx1250 hardware tanh (V_TANH_F32) when the builtin exists; libm fallback.
#if __has_builtin(__builtin_amdgcn_tanhf)
  #define TANHF(x) __builtin_amdgcn_tanhf(x)
#elif __has_builtin(__builtin_amdgcn_tanh_f32)
  #define TANHF(x) __builtin_amdgcn_tanh_f32(x)
#else
  #define TANHF(x) tanhf(x)
#endif

__device__ __forceinline__ unsigned short f2bf(float f) {
    unsigned int u = __float_as_uint(f);
    u += 0x7FFFu + ((u >> 16) & 1u);          // round-to-nearest-even
    return (unsigned short)(u >> 16);
}
__device__ __forceinline__ float bf2f(unsigned short s) {
    return __uint_as_float(((unsigned int)s) << 16);
}
__device__ __forceinline__ v8f zero8() {
    v8f z = {0.f, 0.f, 0.f, 0.f, 0.f, 0.f, 0.f, 0.f};
    return z;
}

// A-fragment (16x32 bf16, MxK) from LDS, row-major with pitch `pitch` halfwords.
// lanes 0-15: row M=lane, K = kk+0..7 then kk+16..23
// lanes16-31: row M=lane-16, K = kk+8..15 then kk+24..31
__device__ __forceinline__ v16bf afrag_lds(const unsigned short* base, int pitch,
                                           int kk, int lane) {
    int m  = lane & 15;
    int kh = (lane & 16) ? 8 : 0;
    const unsigned short* p = base + m * pitch + kk + kh;
    V16BF r;
    r.u[0] = *(const v4u*)(p);
    r.u[1] = *(const v4u*)(p + 16);
    return r.v;
}

// B-fragment (32x16 bf16, KxN) from global memory stored transposed (N x K,
// K contiguous with pitch `pitchK` halfwords).
// lanes 0-15: col N=nbase+lane, K = kk+0..7 then kk+16..23 ; lanes 16-31: K = kk+8..15 then kk+24..31
__device__ __forceinline__ v16bf bfrag_glb(const unsigned short* __restrict__ Bt,
                                           int pitchK, int nbase, int kk, int lane) {
    int n  = nbase + (lane & 15);
    int k0 = kk + ((lane & 16) ? 8 : 0);
    const unsigned short* p = Bt + n * pitchK + k0;
    V16BF r;
    r.u[0] = *(const v4u*)(p);
    r.u[1] = *(const v4u*)(p + 16);
    return r.v;
}

#define WMMA_BF16(a, b, c) \
    __builtin_amdgcn_wmma_f32_16x16x32_bf16(false, (a), false, (b), (short)0, (c), false, false)

// ---------------------------------------------------------------------------
// Prep kernels: bf16 weight transposes + the divergence kernel matrix
//   C[a,b] = W2[a,b] * sum_j W3[b,j] * W1[j,a]   (stored row-major a*512+b,
//   which is exactly the N x K layout needed for the y = g2 @ C^T GEMM).
// ---------------------------------------------------------------------------
__global__ void prep_w1(const float* __restrict__ W1, unsigned short* __restrict__ W1T) {
    int i = blockIdx.x * 256 + threadIdx.x;          // 512 * 64
    if (i >= HID * 64) return;
    int n = i >> 6, k = i & 63;                      // layout: n*64 + k
    float v = (k < DIN + 1) ? W1[k * HID + n] : 0.0f;
    W1T[i] = f2bf(v);
}
__global__ void prep_w2(const float* __restrict__ W2, unsigned short* __restrict__ W2T) {
    int i = blockIdx.x * 256 + threadIdx.x;          // 512 * 512
    if (i >= HID * HID) return;
    int n = i >> 9, k = i & 511;                     // layout: n*512 + k
    W2T[i] = f2bf(W2[k * HID + n]);
}
__global__ void prep_w3(const float* __restrict__ W3, unsigned short* __restrict__ W3T) {
    int i = blockIdx.x * 256 + threadIdx.x;          // 32 * 512
    if (i >= DIN * HID) return;
    int n = i >> 9, k = i & 511;                     // layout: n*512 + k
    W3T[i] = f2bf(W3[k * DIN + n]);
}
__global__ void prep_C(const float* __restrict__ W1, const float* __restrict__ W2,
                       const float* __restrict__ W3, unsigned short* __restrict__ Cbf) {
    int i = blockIdx.x * 256 + threadIdx.x;          // 512 * 512
    if (i >= HID * HID) return;
    int a = i >> 9, b = i & 511;
    float s = 0.0f;
    #pragma unroll
    for (int j = 0; j < DIN; ++j) s += W3[b * DIN + j] * W1[j * HID + a];
    Cbf[i] = f2bf(W2[a * HID + b] * s);              // layout: a*512 + b
}

// ---------------------------------------------------------------------------
// Main kernel: 16 batch rows per WG, 8 wave32s; bf16 WMMA everywhere.
// ---------------------------------------------------------------------------
__global__ __launch_bounds__(256) void cnf_main(
    const float* __restrict__ tptr, const float* __restrict__ x,
    const float* __restrict__ b1, const float* __restrict__ b2,
    const float* __restrict__ b3,
    const unsigned short* __restrict__ W1T, const unsigned short* __restrict__ W2T,
    const unsigned short* __restrict__ W3T, const unsigned short* __restrict__ Cbf,
    float* __restrict__ out) {

    __shared__ __attribute__((aligned(16))) unsigned short sInp[MT * PI];  // padded input (K=64)
    __shared__ __attribute__((aligned(16))) unsigned short sH  [MT * PH];  // h1, then h2
    __shared__ __attribute__((aligned(16))) unsigned short sG1 [MT * PH];  // 1 - h1^2
    __shared__ __attribute__((aligned(16))) unsigned short sG2 [MT * PH];  // 1 - h2^2
    __shared__ float sDiv[MT];

    const int tid  = threadIdx.x;
    const int wave = tid >> 5;
    const int lane = tid & 31;
    const int ncol = lane & 15;
    const int hi8  = (lane & 16) ? 8 : 0;
    const int row0 = blockIdx.x * MT;
    const float tval = tptr[0];

    // ---- stage 0: stage input tile [xs | t | 0-pad] into LDS as bf16 ----
    for (int i = tid; i < MT * 64; i += 256) {
        int r = i >> 6, k = i & 63;
        float v = 0.0f;
        if (k < DIN)           v = x[(row0 + r) * (DIN + 1) + k];
        else if (k == DIN)     v = tval;
        sInp[r * PI + k] = f2bf(v);
    }
    if (tid < MT) sDiv[tid] = 0.0f;
    __syncthreads();

    // ---- stage 1: h1 = tanh(inp @ W1 + b1), g1 = 1 - h1^2  (K=64, N=512) ----
    {
        const int nbase = wave * 64;
        v8f acc[4];
        #pragma unroll
        for (int t = 0; t < 4; ++t) acc[t] = zero8();
        #pragma unroll
        for (int kk = 0; kk < 64; kk += 32) {
            v16bf a = afrag_lds(sInp, PI, kk, lane);
            #pragma unroll
            for (int t = 0; t < 4; ++t) {
                v16bf b = bfrag_glb(W1T, 64, nbase + t * 16, kk, lane);
                acc[t] = WMMA_BF16(a, b, acc[t]);
            }
        }
        #pragma unroll
        for (int t = 0; t < 4; ++t) {
            int n = nbase + t * 16 + ncol;
            float bias = b1[n];
            #pragma unroll
            for (int v = 0; v < 8; ++v) {
                int m = v + hi8;
                float h = TANHF(acc[t][v] + bias);
                sH [m * PH + n] = f2bf(h);
                sG1[m * PH + n] = f2bf(1.0f - h * h);
            }
        }
    }
    __syncthreads();

    // ---- stage 2: h2 = tanh(h1 @ W2 + b2), g2 = 1 - h2^2  (K=512, N=512) ----
    {
        const int nbase = wave * 64;
        v8f acc[4];
        #pragma unroll
        for (int t = 0; t < 4; ++t) acc[t] = zero8();
        for (int kk = 0; kk < HID; kk += 32) {
            v16bf a = afrag_lds(sH, PH, kk, lane);
            #pragma unroll
            for (int t = 0; t < 4; ++t) {
                v16bf b = bfrag_glb(W2T, HID, nbase + t * 16, kk, lane);
                acc[t] = WMMA_BF16(a, b, acc[t]);
            }
        }
        __syncthreads();   // everyone done reading h1 before we overwrite sH with h2
        #pragma unroll
        for (int t = 0; t < 4; ++t) {
            int n = nbase + t * 16 + ncol;
            float bias = b2[n];
            #pragma unroll
            for (int v = 0; v < 8; ++v) {
                int m = v + hi8;
                float h = TANHF(acc[t][v] + bias);
                sH [m * PH + n] = f2bf(h);        // now holds h2
                sG2[m * PH + n] = f2bf(1.0f - h * h);
            }
        }
    }
    __syncthreads();

    // ---- stage 3a: dx = h2 @ W3 + b3  (K=512, N=32) -> waves 0,1 ----
    if (wave < 2) {
        const int nbase = wave * 16;
        v8f acc = zero8();
        for (int kk = 0; kk < HID; kk += 32) {
            v16bf a = afrag_lds(sH, PH, kk, lane);
            v16bf b = bfrag_glb(W3T, HID, nbase, kk, lane);
            acc = WMMA_BF16(a, b, acc);
        }
        int n = nbase + ncol;
        float bias = b3[n];
        #pragma unroll
        for (int v = 0; v < 8; ++v) {
            int m = v + hi8;
            out[(row0 + m) * (DIN + 1) + n] = acc[v] + bias;
        }
    }

    // ---- stage 3b: y = g2 @ C^T ; div_i = <g1_i, y_i>  (K=512, N=512) ----
    {
        const int nbase = wave * 64;
        v8f acc[4];
        #pragma unroll
        for (int t = 0; t < 4; ++t) acc[t] = zero8();
        for (int kk = 0; kk < HID; kk += 32) {
            v16bf a = afrag_lds(sG2, PH, kk, lane);
            #pragma unroll
            for (int t = 0; t < 4; ++t) {
                v16bf b = bfrag_glb(Cbf, HID, nbase + t * 16, kk, lane);
                acc[t] = WMMA_BF16(a, b, acc[t]);
            }
        }
        float pp[8];
        #pragma unroll
        for (int v = 0; v < 8; ++v) pp[v] = 0.0f;
        #pragma unroll
        for (int t = 0; t < 4; ++t) {
            int n = nbase + t * 16 + ncol;
            #pragma unroll
            for (int v = 0; v < 8; ++v) {
                int m = v + hi8;
                pp[v] += acc[t][v] * bf2f(sG1[m * PH + n]);
            }
        }
        #pragma unroll
        for (int v = 0; v < 8; ++v) atomicAdd(&sDiv[v + hi8], pp[v]);
    }
    __syncthreads();

    if (tid < MT) out[(row0 + tid) * (DIN + 1) + DIN] = sDiv[tid];
}

// ---------------------------------------------------------------------------
// Launcher. Workspace layout (bf16 halfwords):
//   [0,      64KB)  W1T  512 x 64   (K padded 33 -> 64)
//   [64KB,   96KB)  W3T   32 x 512
//   [96KB,  608KB)  W2T  512 x 512
//   [608KB,1120KB)  Cbf  512 x 512  (divergence kernel matrix)
// ---------------------------------------------------------------------------
extern "C" void kernel_launch(void* const* d_in, const int* in_sizes, int n_in,
                              void* d_out, int out_size, void* d_ws, size_t ws_size,
                              hipStream_t stream) {
    const float* t  = (const float*)d_in[0];
    const float* x  = (const float*)d_in[1];
    const float* W1 = (const float*)d_in[2];
    const float* b1 = (const float*)d_in[3];
    const float* W2 = (const float*)d_in[4];
    const float* b2 = (const float*)d_in[5];
    const float* W3 = (const float*)d_in[6];
    const float* b3 = (const float*)d_in[7];
    float* out = (float*)d_out;

    unsigned char* ws = (unsigned char*)d_ws;
    unsigned short* W1T = (unsigned short*)(ws);
    unsigned short* W3T = (unsigned short*)(ws + 65536);
    unsigned short* W2T = (unsigned short*)(ws + 65536 + 32768);
    unsigned short* Cbf = (unsigned short*)(ws + 65536 + 32768 + 524288);

    prep_w1<<<(HID * 64 + 255) / 256, 256, 0, stream>>>(W1, W1T);
    prep_w3<<<(DIN * HID + 255) / 256, 256, 0, stream>>>(W3, W3T);
    prep_w2<<<(HID * HID + 255) / 256, 256, 0, stream>>>(W2, W2T);
    prep_C <<<(HID * HID + 255) / 256, 256, 0, stream>>>(W1, W2, W3, Cbf);

    cnf_main<<<BATCH / MT, 256, 0, stream>>>(t, x, b1, b2, b3,
                                             W1T, W2T, W3T, Cbf, out);
}
